// DiffusionPriorNetwork_22883585753250
// MI455X (gfx1250) — matrix operand order repaired and checked by reading, not compiled, not run on hardware
//
#include <hip/hip_runtime.h>
#include <hip/hip_bf16.h>
#include <math.h>

typedef __bf16 bf16;
typedef __attribute__((ext_vector_type(16))) __bf16 v16bf;
typedef __attribute__((ext_vector_type(8)))  __bf16 v8bf;
typedef __attribute__((ext_vector_type(8)))  float  v8f;

#define DIM    768
#define DEPTH  12
#define HEADS  12
#define DH     64
#define FF     3072
#define NTOK   260
#define NTOKP  288   /* padded q rows per (b,h) so async loads never go OOB */
#define NKEY   261
#define NKPAD  272   /* key tiles padded to 17x16 */
#define NKROWS 320   /* allocated k/v rows so 64-row async chunks never go OOB */
#define BATCH  128
#define TTOK   (BATCH * NTOK)
#define SCW    296   /* fp32 score row stride */
#define SBW    304   /* bf16 prob overlay row stride (16B aligned) */

static __device__ __forceinline__ v16bf cat8(v8bf lo, v8bf hi) {
  return __builtin_shufflevector(lo, hi, 0,1,2,3,4,5,6,7,8,9,10,11,12,13,14,15);
}
static __device__ __forceinline__ v8f wmma_bf16(v16bf a, v16bf b, v8f c) {
  return __builtin_amdgcn_wmma_f32_16x16x32_bf16(false, a, false, b, (short)0, c, false, false);
}
static __device__ __forceinline__ float fast_rcp(float x) {
#if __has_builtin(__builtin_amdgcn_rcpf)
  return __builtin_amdgcn_rcpf(x);
#else
  return 1.f / x;
#endif
}
// CDNA5 async global->LDS copy (16B per lane), tracked by ASYNCcnt.
static __device__ __forceinline__ void async_ld_b128(unsigned ldsOff, const void* g) {
  asm volatile("global_load_async_to_lds_b128 %0, %1, off"
               :: "v"(ldsOff), "v"((unsigned long long)(size_t)g) : "memory");
}
static __device__ __forceinline__ void wait_async0() {
#if __has_builtin(__builtin_amdgcn_s_wait_asynccnt)
  __builtin_amdgcn_s_wait_asynccnt(0);
#else
  asm volatile("s_wait_asynccnt 0x0" ::: "memory");
#endif
}
static __device__ __forceinline__ unsigned lds_off(const void* p) {
  return (unsigned)(size_t)p;  // low 32 bits of flat LDS address = LDS byte offset
}

// ---------------- fp32 [K][N] -> bf16 transposed [N][K] ----------
// grid (K/32, N/32, layers); block 256 (32x8). Output is K-contiguous per
// output column so GEMM B-tiles can be staged with async b128 directly.
__global__ __launch_bounds__(256) void tconv_k(const float* __restrict__ in,
                                               bf16* __restrict__ out,
                                               int K, int N) {
  __shared__ float t[32][33];
  int k0 = blockIdx.x * 32, n0 = blockIdx.y * 32;
  const float* src = in + (size_t)blockIdx.z * K * N;
  bf16* dst = out + (size_t)blockIdx.z * K * N;
  int tx = threadIdx.x & 31, ty = threadIdx.x >> 5;
#pragma unroll
  for (int i = 0; i < 4; i++) {
    int kk = ty + i * 8;
    t[kk][tx] = src[(size_t)(k0 + kk) * N + n0 + tx];
  }
  __syncthreads();
#pragma unroll
  for (int i = 0; i < 4; i++) {
    int nn = ty + i * 8;
    dst[(size_t)(n0 + nn) * K + k0 + tx] = (bf16)t[tx][nn];
  }
}

// ---------------- build x ----------------------------------------
__global__ __launch_bounds__(256) void build_x_k(
    const float* __restrict__ text_enc, const float* __restrict__ null_text,
    const float* __restrict__ text_embed, const float* __restrict__ time_emb,
    const int* __restrict__ tsteps, const float* __restrict__ image_embed,
    const float* __restrict__ learned_q, float* __restrict__ x) {
  int n = blockIdx.x;  // 0..259
  int b = blockIdx.y;
  float* dst = x + ((size_t)b * NTOK + n) * DIM;
  if (n < 256) {
    const float* src = text_enc + ((size_t)b * 256 + n) * DIM;
    __shared__ int anyf;
    if (threadIdx.x == 0) anyf = 0;
    __syncthreads();
    float v[3]; int loc = 0;
    for (int i = 0; i < 3; i++) { v[i] = src[threadIdx.x + i * 256]; loc |= (v[i] != 0.f); }
    if (loc) anyf = 1;
    __syncthreads();
    if (anyf) {
      for (int i = 0; i < 3; i++) dst[threadIdx.x + i * 256] = v[i];
    } else {
      const float* ns = null_text + (size_t)n * DIM;
      for (int i = 0; i < 3; i++) { int d = threadIdx.x + i * 256; dst[d] = ns[d]; }
    }
  } else {
    const float* src;
    if (n == 256)      src = text_embed + (size_t)b * DIM;
    else if (n == 257) src = time_emb + (size_t)tsteps[b] * DIM;
    else if (n == 258) src = image_embed + (size_t)b * DIM;
    else               src = learned_q;
    for (int i = 0; i < 3; i++) { int d = threadIdx.x + i * 256; dst[d] = src[d]; }
  }
}

// ---------------- relative position bias table -------------------
__global__ void bias_k(const float* __restrict__ rel_bias, float* __restrict__ biasb) {
  int i = blockIdx.x, h = blockIdx.y;
  int j = threadIdx.x;
  if (j >= NKEY) return;
  int nn = i - j; if (nn < 0) nn = 0;
  int bucket;
  if (nn < 16) bucket = nn;
  else {
    float lv = logf((float)nn / 16.f) / logf(8.f) * 16.f;
    bucket = 16 + (int)lv;
    if (bucket > 31) bucket = 31;
  }
  biasb[((size_t)h * NTOK + i) * NKEY + j] = rel_bias[bucket * HEADS + h];
}

// ---------------- rotary tables ----------------------------------
__global__ void rot_k(float* __restrict__ cosb, float* __restrict__ sinb) {
  int n = blockIdx.x; int d = threadIdx.x;  // 32 threads
  int p = d >> 1;
  float invf = powf(10000.f, -(float)p / 16.f);
  float f = (float)n * invf;
  cosb[n * 32 + d] = cosf(f);
  sinb[n * 32 + d] = sinf(f);
}

// ---------------- LayerNorm -> bf16 ------------------------------
__global__ __launch_bounds__(256) void ln_k(const float* __restrict__ x,
                                            const float* __restrict__ g,
                                            bf16* __restrict__ y) {
  int row = blockIdx.x;
  const float* xr = x + (size_t)row * DIM;
  __shared__ float s1[8], s2[8];
  float v[3]; float sum = 0.f, sq = 0.f;
  for (int i = 0; i < 3; i++) { v[i] = xr[threadIdx.x + i * 256]; sum += v[i]; sq += v[i] * v[i]; }
  for (int o = 16; o; o >>= 1) { sum += __shfl_xor(sum, o); sq += __shfl_xor(sq, o); }
  if ((threadIdx.x & 31) == 0) { s1[threadIdx.x >> 5] = sum; s2[threadIdx.x >> 5] = sq; }
  __syncthreads();
  sum = 0.f; sq = 0.f;
  for (int i = 0; i < 8; i++) { sum += s1[i]; sq += s2[i]; }
  float mean = sum * (1.f / DIM);
  float var = sq * (1.f / DIM) - mean * mean;
  float inv = rsqrtf(var + 1e-5f);
  for (int i = 0; i < 3; i++) {
    int d = threadIdx.x + i * 256;
    y[(size_t)row * DIM + d] = (bf16)((v[i] - mean) * inv * g[d]);
  }
}

// ---------------- LayerNorm(tmp)*g + residual into x -------------
__global__ __launch_bounds__(256) void ln_res_k(const float* __restrict__ t,
                                                const float* __restrict__ g,
                                                float* __restrict__ x) {
  int row = blockIdx.x;
  const float* tr = t + (size_t)row * DIM;
  float* xr = x + (size_t)row * DIM;
  __shared__ float s1[8], s2[8];
  float v[3]; float sum = 0.f, sq = 0.f;
  for (int i = 0; i < 3; i++) { v[i] = tr[threadIdx.x + i * 256]; sum += v[i]; sq += v[i] * v[i]; }
  for (int o = 16; o; o >>= 1) { sum += __shfl_xor(sum, o); sq += __shfl_xor(sq, o); }
  if ((threadIdx.x & 31) == 0) { s1[threadIdx.x >> 5] = sum; s2[threadIdx.x >> 5] = sq; }
  __syncthreads();
  sum = 0.f; sq = 0.f;
  for (int i = 0; i < 8; i++) { sum += s1[i]; sq += s2[i]; }
  float mean = sum * (1.f / DIM);
  float var = sq * (1.f / DIM) - mean * mean;
  float inv = rsqrtf(var + 1e-5f);
  for (int i = 0; i < 3; i++) {
    int d = threadIdx.x + i * 256;
    xr[d] += (v[i] - mean) * inv * g[d];
  }
}

// ---------------- stable LayerNorm on last token -> bf16 ---------
__global__ __launch_bounds__(256) void ln_last_k(const float* __restrict__ x,
                                                 const float* __restrict__ g,
                                                 bf16* __restrict__ y) {
  int bb = blockIdx.x;
  const float* xr = x + ((size_t)bb * NTOK + (NTOK - 1)) * DIM;
  __shared__ float s1[8], s2[8], s3[8];
  float v[3]; float mx = -3.4e38f;
  for (int i = 0; i < 3; i++) { v[i] = xr[threadIdx.x + i * 256]; mx = fmaxf(mx, v[i]); }
  for (int o = 16; o; o >>= 1) mx = fmaxf(mx, __shfl_xor(mx, o));
  if ((threadIdx.x & 31) == 0) s3[threadIdx.x >> 5] = mx;
  __syncthreads();
  mx = s3[0];
  for (int i = 1; i < 8; i++) mx = fmaxf(mx, s3[i]);
  float sum = 0.f, sq = 0.f;
  for (int i = 0; i < 3; i++) { v[i] /= mx; sum += v[i]; sq += v[i] * v[i]; }
  for (int o = 16; o; o >>= 1) { sum += __shfl_xor(sum, o); sq += __shfl_xor(sq, o); }
  if ((threadIdx.x & 31) == 0) { s1[threadIdx.x >> 5] = sum; s2[threadIdx.x >> 5] = sq; }
  __syncthreads();
  sum = 0.f; sq = 0.f;
  for (int i = 0; i < 8; i++) { sum += s1[i]; sq += s2[i]; }
  float mean = sum * (1.f / DIM);
  float var = sq * (1.f / DIM) - mean * mean;
  float inv = rsqrtf(var + 1e-5f);
  for (int i = 0; i < 3; i++) {
    int d = threadIdx.x + i * 256;
    y[(size_t)bb * DIM + d] = (bf16)((v[i] - mean) * inv * g[d]);
  }
}

// ---------------- residual add -----------------------------------
__global__ void add_res_k(float* __restrict__ x, const float* __restrict__ t, size_t count) {
  size_t i = (size_t)blockIdx.x * blockDim.x + threadIdx.x;
  size_t stride = (size_t)gridDim.x * blockDim.x;
  for (; i < count; i += stride) x[i] += t[i];
}

// ---------------- q post: rotary + l2norm*4 -> bf16 (b,h,n,d) ----
__global__ __launch_bounds__(256) void qpost_k(const float* __restrict__ qf,
                                               const float* __restrict__ cosb,
                                               const float* __restrict__ sinb,
                                               bf16* __restrict__ qbf) {
  int t = blockIdx.x;
  int b = t / NTOKP, n = t % NTOKP;
  int wave = threadIdx.x >> 5, lane = threadIdx.x & 31;
  for (int h = wave; h < HEADS; h += 8) {
    bf16* dst = qbf + (((size_t)b * HEADS + h) * NTOKP + n) * DH;
    if (n >= NTOK) { dst[lane] = (bf16)0.f; dst[lane + 32] = (bf16)0.f; continue; }
    const float* src = qf + ((size_t)b * NTOK + n) * DIM + h * DH;
    float x0 = src[lane];
    float x1 = src[lane + 32];
    float xp = __shfl_xor(x0, 1);
    float rh = (lane & 1) ? xp : -xp;
    float r0 = x0 * cosb[n * 32 + lane] + rh * sinb[n * 32 + lane];
    float ss = r0 * r0 + x1 * x1;
    for (int o = 16; o; o >>= 1) ss += __shfl_xor(ss, o);
    float inv = 4.0f / fmaxf(sqrtf(ss), 1e-12f);  // x16 pre-scale cancels in l2norm
    dst[lane] = (bf16)(r0 * inv);
    dst[lane + 32] = (bf16)(x1 * inv);
  }
}

// ---------------- k/v post --------------------------------------
// k -> kbf[b][j][d] (row-major); v -> vt[b][d][j] (transposed, j-contiguous)
__global__ __launch_bounds__(32) void kvpost_k(const float* __restrict__ kvf,
                                               const float* __restrict__ nullkv,
                                               const float* __restrict__ cosb,
                                               const float* __restrict__ sinb,
                                               bf16* __restrict__ kbf, bf16* __restrict__ vt) {
  int idx = blockIdx.x; int b = idx / NKROWS; int j = idx % NKROWS;
  int lane = threadIdx.x;
  float k0, k1, v0, v1;
  if (j == 0) {
    k0 = nullkv[lane]; k1 = nullkv[lane + 32];
    v0 = nullkv[64 + lane]; v1 = nullkv[96 + lane];
  } else if (j <= NTOK) {
    int n = j - 1;
    const float* src = kvf + ((size_t)b * NTOK + n) * 128;
    float x0 = src[lane], x1 = src[lane + 32];
    float xp = __shfl_xor(x0, 1);
    float rh = (lane & 1) ? xp : -xp;
    k0 = x0 * cosb[n * 32 + lane] + rh * sinb[n * 32 + lane];
    k1 = x1;
    v0 = src[64 + lane]; v1 = src[96 + lane];
  } else {
    k0 = k1 = v0 = v1 = 0.f;
  }
  float ss = k0 * k0 + k1 * k1;
  for (int o = 16; o; o >>= 1) ss += __shfl_xor(ss, o);
  float inv = (j <= NTOK) ? 4.0f / fmaxf(sqrtf(ss), 1e-12f) : 0.f;
  bf16* kd = kbf + ((size_t)b * NKROWS + j) * DH;
  kd[lane] = (bf16)(k0 * inv); kd[lane + 32] = (bf16)(k1 * inv);
  vt[((size_t)b * DH + lane) * NKROWS + j]      = (bf16)v0;
  vt[((size_t)b * DH + lane + 32) * NKROWS + j] = (bf16)v1;
}

// ---------------- tiled bf16 WMMA GEMM, fully async staging ------
// C[MxN] = A[MxK] * Bt[NxK]^T; Bt is the pre-transposed weight (K-contiguous).
// block tile 128x64, 8 waves, wave tile 32x32; double-buffered LDS.
template <int OUT_BF16>
__global__ __launch_bounds__(256) void gemm_k(const bf16* __restrict__ A,
                                              const bf16* __restrict__ Bt,
                                              void* __restrict__ C,
                                              int M, int N, int K,
                                              int lda, int ldb, int ldc) {
  __shared__ __align__(16) bf16 As[2][128][40];
  __shared__ __align__(16) bf16 Bs[2][64][40];
  int tid = threadIdx.x, lane = tid & 31, wave = tid >> 5;
  int half8 = (lane >> 4) * 8;
  int mrow = lane & 15;
  int waveM = wave >> 1, waveN = wave & 1;
  int mBase = blockIdx.y * 128, nBase = blockIdx.x * 64;
  v8f acc[2][2] = {};

  auto issueA = [&](int k0, int buf) {
#pragma unroll
    for (int it = 0; it < 2; ++it) {
      int c = tid + it * 256;               // 512 chunks of 16B
      int r = c >> 2, cw = (c & 3) * 8;
      async_ld_b128(lds_off(&As[buf][r][cw]),
                    A + (size_t)(mBase + r) * lda + k0 + cw);
    }
  };
  auto issueB = [&](int k0, int buf) {
    int r = tid >> 2, cw = (tid & 3) * 8;   // 256 chunks of 16B
    async_ld_b128(lds_off(&Bs[buf][r][cw]),
                  Bt + (size_t)(nBase + r) * ldb + k0 + cw);
  };

  issueA(0, 0);
  issueB(0, 0);
  wait_async0();
  __syncthreads();

  int buf = 0;
  for (int k0 = 0; k0 < K; k0 += 32, buf ^= 1) {
    bool more = (k0 + 32) < K;
    if (more) { issueA(k0 + 32, buf ^ 1); issueB(k0 + 32, buf ^ 1); }
    v16bf af[2], bfv[2];
#pragma unroll
    for (int mt = 0; mt < 2; mt++) {
      int m = waveM * 32 + mt * 16 + mrow;
      af[mt] = cat8(*(const v8bf*)&As[buf][m][half8],
                    *(const v8bf*)&As[buf][m][16 + half8]);
    }
#pragma unroll
    for (int nt = 0; nt < 2; nt++) {
      int n = waveN * 32 + nt * 16 + mrow;
      bfv[nt] = cat8(*(const v8bf*)&Bs[buf][n][half8],
                     *(const v8bf*)&Bs[buf][n][16 + half8]);
    }
#pragma unroll
    for (int mt = 0; mt < 2; mt++)
#pragma unroll
      for (int nt = 0; nt < 2; nt++)
        acc[mt][nt] = wmma_bf16(af[mt], bfv[nt], acc[mt][nt]);
    if (more) wait_async0();
    __syncthreads();
  }

  int rOff = (lane >> 4) * 8;
  for (int mt = 0; mt < 2; mt++)
    for (int nt = 0; nt < 2; nt++)
      for (int r = 0; r < 8; r++) {
        int m = mBase + waveM * 32 + mt * 16 + rOff + r;
        int n = nBase + waveN * 32 + nt * 16 + mrow;
        if (m < M) {
          size_t off = (size_t)m * ldc + n;
          if (OUT_BF16) ((bf16*)C)[off] = (bf16)acc[mt][nt][r];
          else          ((float*)C)[off] = acc[mt][nt][r];
        }
      }
}

// ---------------- gated GEMM for ff_w1: out = a * silu(g) --------
// Bt is w1 pre-transposed: [6144][768]; rows n -> a-half, rows n+3072 -> gate.
__global__ __launch_bounds__(256) void gemm_gated_k(const bf16* __restrict__ A,
                                                    const bf16* __restrict__ Bt,
                                                    bf16* __restrict__ O,
                                                    int M, int K, int lda, int ldb) {
  __shared__ __align__(16) bf16 As[2][128][40];
  __shared__ __align__(16) bf16 Bs[2][64][40];
  __shared__ __align__(16) bf16 Gs[2][64][40];
  int tid = threadIdx.x, lane = tid & 31, wave = tid >> 5;
  int half8 = (lane >> 4) * 8;
  int mrow = lane & 15;
  int waveM = wave >> 1, waveN = wave & 1;
  int mBase = blockIdx.y * 128, nBase = blockIdx.x * 64;
  v8f accA[2][2] = {};
  v8f accG[2][2] = {};

  auto issueA = [&](int k0, int buf) {
#pragma unroll
    for (int it = 0; it < 2; ++it) {
      int c = tid + it * 256;
      int r = c >> 2, cw = (c & 3) * 8;
      async_ld_b128(lds_off(&As[buf][r][cw]),
                    A + (size_t)(mBase + r) * lda + k0 + cw);
    }
  };
  auto issueBG = [&](int k0, int buf) {
    int r = tid >> 2, cw = (tid & 3) * 8;
    async_ld_b128(lds_off(&Bs[buf][r][cw]),
                  Bt + (size_t)(nBase + r) * ldb + k0 + cw);
    async_ld_b128(lds_off(&Gs[buf][r][cw]),
                  Bt + (size_t)(nBase + FF + r) * ldb + k0 + cw);
  };

  issueA(0, 0);
  issueBG(0, 0);
  wait_async0();
  __syncthreads();

  int buf = 0;
  for (int k0 = 0; k0 < K; k0 += 32, buf ^= 1) {
    bool more = (k0 + 32) < K;
    if (more) { issueA(k0 + 32, buf ^ 1); issueBG(k0 + 32, buf ^ 1); }
    v16bf af[2], bfv[2], gfv[2];
#pragma unroll
    for (int mt = 0; mt < 2; mt++) {
      int m = waveM * 32 + mt * 16 + mrow;
      af[mt] = cat8(*(const v8bf*)&As[buf][m][half8],
                    *(const v8bf*)&As[buf][m][16 + half8]);
    }
#pragma unroll
    for (int nt = 0; nt < 2; nt++) {
      int n = waveN * 32 + nt * 16 + mrow;
      bfv[nt] = cat8(*(const v8bf*)&Bs[buf][n][half8],
                     *(const v8bf*)&Bs[buf][n][16 + half8]);
      gfv[nt] = cat8(*(const v8bf*)&Gs[buf][n][half8],
                     *(const v8bf*)&Gs[buf][n][16 + half8]);
    }
#pragma unroll
    for (int mt = 0; mt < 2; mt++)
#pragma unroll
      for (int nt = 0; nt < 2; nt++) {
        accA[mt][nt] = wmma_bf16(af[mt], bfv[nt], accA[mt][nt]);
        accG[mt][nt] = wmma_bf16(af[mt], gfv[nt], accG[mt][nt]);
      }
    if (more) wait_async0();
    __syncthreads();
  }

  int rOff = (lane >> 4) * 8;
  for (int mt = 0; mt < 2; mt++)
    for (int nt = 0; nt < 2; nt++)
      for (int r = 0; r < 8; r++) {
        int m = mBase + waveM * 32 + mt * 16 + rOff + r;
        int n = nBase + waveN * 32 + nt * 16 + mrow;
        if (m < M) {
          float a = accA[mt][nt][r];
          float g = accG[mt][nt][r];
          float sg = g * fast_rcp(1.f + __expf(-g));  // silu via v_rcp_f32
          O[(size_t)m * FF + n] = (bf16)(a * sg);
        }
      }
}

// ---------------- fused attention --------------------------------
// grid: (9 q-chunks of 32, HEADS, BATCH); block 256 = 8 waves.
__global__ __launch_bounds__(256) void attn_k(const bf16* __restrict__ qbf,
                                              const bf16* __restrict__ kbf,
                                              const bf16* __restrict__ vt,
                                              const float* __restrict__ biasb,
                                              bf16* __restrict__ aout) {
  int qc = blockIdx.x, h = blockIdx.y, b = blockIdx.z;
  int qbase = qc * 32;
  __shared__ __align__(16) bf16 qs[32][72];
  __shared__ __align__(16) bf16 kvs[64][72];
  __shared__ __align__(32) float sc[32][SCW];
  __shared__ float red[32][8];
  bf16* scb = (bf16*)&sc[0][0];  // bf16 prob overlay, row stride SBW
  int tid = threadIdx.x, lane = tid & 31, wave = tid >> 5;
  int tileM = wave >> 2, tileN = wave & 3;
  int half8 = (lane >> 4) * 8;
  int rOff = (lane >> 4) * 8;
  int mrow = lane & 15;

  // async-stage q tile (contiguous 16B chunks)
  {
    int r = tid >> 3, cw = (tid & 7) * 8;
    async_ld_b128(lds_off(&qs[r][cw]),
                  qbf + (((size_t)b * HEADS + h) * NTOKP + qbase + r) * DH + cw);
  }

  // phase 1: scores = q @ k^T  (K-dim = 64 head dims)
  for (int jc = 0; jc < NKPAD; jc += 64) {
    __syncthreads();  // all waves done reading previous kvs chunk
#pragma unroll
    for (int it = 0; it < 2; ++it) {
      int c = tid + it * 256;               // 512 chunks of 16B
      int r = c >> 3, cw = (c & 7) * 8;
      async_ld_b128(lds_off(&kvs[r][cw]),
                    kbf + ((size_t)b * NKROWS + jc + r) * DH + cw);
    }
    wait_async0();
    __syncthreads();
    v8f acc = {};
#pragma unroll
    for (int kk = 0; kk < 64; kk += 32) {
      int m = tileM * 16 + mrow;
      v16bf af = cat8(*(const v8bf*)&qs[m][kk + half8],
                      *(const v8bf*)&qs[m][kk + 16 + half8]);
      int jl = tileN * 16 + mrow;
      v16bf bfv = cat8(*(const v8bf*)&kvs[jl][kk + half8],
                       *(const v8bf*)&kvs[jl][kk + 16 + half8]);
      acc = wmma_bf16(af, bfv, acc);
    }
    for (int r = 0; r < 8; r++) {
      int rr = tileM * 16 + rOff + r;
      int iq = qbase + rr;
      int j = jc + tileN * 16 + mrow;
      if (j < NKPAD) {
        float s;
        if (iq >= NTOK || j > NTOK || j >= iq + 2) s = -3e38f;  // causal(k=2) + pads
        else s = acc[r] + biasb[((size_t)h * NTOK + iq) * NKEY + j];
        sc[rr][j] = s;
      }
    }
  }
  __syncthreads();

  // softmax in fp32; probs written as bf16 overlay (registers carry values
  // across the barrier so fp32 reads all complete before bf16 writes begin)
  {
    int row = tid >> 3, sub = tid & 7;
    float ev[34];
    float mx = -3.4e38f;
    int cnt = 0;
    for (int c = sub; c < NKPAD; c += 8, cnt++) { ev[cnt] = sc[row][c]; mx = fmaxf(mx, ev[cnt]); }
    red[row][sub] = mx;
    __syncthreads();
    mx = red[row][0];
    for (int i = 1; i < 8; i++) mx = fmaxf(mx, red[row][i]);
    float sum = 0.f;
    for (int i = 0; i < cnt; i++) { ev[i] = __expf(ev[i] - mx); sum += ev[i]; }
    __syncthreads();
    red[row][sub] = sum;
    __syncthreads();
    sum = 0.f;
    for (int i = 0; i < 8; i++) sum += red[row][i];
    float inv = 1.0f / sum;
    int idx = 0;
    for (int c = sub; c < SBW; c += 8, idx++)
      scb[row * SBW + c] = (c < NKPAD) ? (bf16)(ev[idx] * inv) : (bf16)0.f;
  }
  __syncthreads();

  // phase 2: out = probs @ v  (K-dim = 272 keys); v staged async from vt
  v8f oacc = {};
  for (int jc = 0; jc < NKPAD; jc += 64) {
    __syncthreads();
#pragma unroll
    for (int it = 0; it < 2; ++it) {
      int c = tid + it * 256;               // 512 chunks of 16B
      int r = c >> 3, cw = (c & 7) * 8;     // r = head dim d, cw = key offset
      async_ld_b128(lds_off(&kvs[r][cw]),
                    vt + ((size_t)b * DH + r) * NKROWS + jc + cw);
    }
    wait_async0();
    __syncthreads();
    for (int kk = 0; kk < 64 && (jc + kk) < NKPAD; kk += 32) {
      int jj = jc + kk;
      const bf16* ap = scb + (tileM * 16 + mrow) * SBW;
      v16bf af = cat8(*(const v8bf*)(ap + jj + half8),
                      *(const v8bf*)(ap + jj + 16 + half8));
      int d = tileN * 16 + mrow;
      v16bf bfv = cat8(*(const v8bf*)&kvs[d][kk + half8],
                       *(const v8bf*)&kvs[d][kk + 16 + half8]);
      oacc = wmma_bf16(af, bfv, oacc);
    }
  }
  for (int r = 0; r < 8; r++) {
    int n = qbase + tileM * 16 + rOff + r;
    if (n < NTOK) {
      int col = h * DH + tileN * 16 + mrow;
      aout[((size_t)b * NTOK + n) * DIM + col] = (bf16)oacc[r];
    }
  }
}

// =================================================================
extern "C" void kernel_launch(void* const* d_in, const int* in_sizes, int n_in,
                              void* d_out, int out_size, void* d_ws, size_t ws_size,
                              hipStream_t stream) {
  const float* image_embed = (const float*)d_in[0];
  const int*   tsteps      = (const int*)d_in[1];
  const float* text_embed  = (const float*)d_in[2];
  const float* text_enc    = (const float*)d_in[3];
  const float* rel_bias    = (const float*)d_in[4];
  const float* time_emb    = (const float*)d_in[5];
  const float* learned_q   = (const float*)d_in[6];
  const float* null_text   = (const float*)d_in[7];
  const float* attn_g      = (const float*)d_in[8];
  const float* to_q        = (const float*)d_in[9];
  const float* to_kv       = (const float*)d_in[10];
  const float* null_kv     = (const float*)d_in[11];
  const float* to_out      = (const float*)d_in[12];
  const float* out_g       = (const float*)d_in[13];
  const float* ff_g        = (const float*)d_in[14];
  const float* ff_w1       = (const float*)d_in[15];
  const float* ff_w2       = (const float*)d_in[16];
  const float* fin_g       = (const float*)d_in[17];
  const float* proj        = (const float*)d_in[18];

  char* p = (char*)d_ws;
  auto carve = [&](size_t bytes) {
    void* r = (void*)p;
    p += (bytes + 255) & ~(size_t)255;
    return r;
  };
  // all weight copies are bf16 AND transposed: Wt[n][k], K-contiguous
  bf16*  wq_t   = (bf16*)carve((size_t)DEPTH * DIM * DIM * 2);
  bf16*  wkv_t  = (bf16*)carve((size_t)DEPTH * DIM * 128 * 2);
  bf16*  wo_t   = (bf16*)carve((size_t)DEPTH * DIM * DIM * 2);
  bf16*  w1_t   = (bf16*)carve((size_t)DEPTH * DIM * 2 * FF * 2);
  bf16*  w2_t   = (bf16*)carve((size_t)DEPTH * FF * DIM * 2);
  bf16*  proj_t = (bf16*)carve((size_t)DIM * DIM * 2);
  float* xf     = (float*)carve((size_t)TTOK * DIM * 4);
  bf16*  hbf    = (bf16*)carve((size_t)TTOK * DIM * 2);
  float* tmpf   = (float*)carve((size_t)TTOK * DIM * 4);
  float* kvf    = (float*)carve((size_t)TTOK * 128 * 4);
  bf16*  qbf    = (bf16*)carve((size_t)BATCH * HEADS * NTOKP * DH * 2);
  bf16*  kbf    = (bf16*)carve((size_t)BATCH * NKROWS * DH * 2);
  bf16*  vtb    = (bf16*)carve((size_t)BATCH * DH * NKROWS * 2);
  bf16*  aobf   = (bf16*)carve((size_t)TTOK * DIM * 2);
  bf16*  gbf    = (bf16*)carve((size_t)TTOK * FF * 2);
  float* biasb  = (float*)carve((size_t)HEADS * NTOK * NKEY * 4);
  float* cosb   = (float*)carve((size_t)NTOK * 32 * 4);
  float* sinb   = (float*)carve((size_t)NTOK * 32 * 4);

  tconv_k<<<dim3(DIM/32, DIM/32, DEPTH), 256, 0, stream>>>(to_q,   wq_t,   DIM, DIM);
  tconv_k<<<dim3(DIM/32, 128/32, DEPTH), 256, 0, stream>>>(to_kv,  wkv_t,  DIM, 128);
  tconv_k<<<dim3(DIM/32, DIM/32, DEPTH), 256, 0, stream>>>(to_out, wo_t,   DIM, DIM);
  tconv_k<<<dim3(DIM/32, (2*FF)/32, DEPTH), 256, 0, stream>>>(ff_w1, w1_t, DIM, 2*FF);
  tconv_k<<<dim3(FF/32, DIM/32, DEPTH), 256, 0, stream>>>(ff_w2,  w2_t,   FF, DIM);
  tconv_k<<<dim3(DIM/32, DIM/32, 1), 256, 0, stream>>>(proj,      proj_t, DIM, DIM);

  build_x_k<<<dim3(NTOK, BATCH), 256, 0, stream>>>(text_enc, null_text, text_embed,
                                                   time_emb, tsteps, image_embed,
                                                   learned_q, xf);
  bias_k<<<dim3(NTOK, HEADS), 288, 0, stream>>>(rel_bias, biasb);
  rot_k<<<NTOK, 32, 0, stream>>>(cosb, sinb);

  for (int l = 0; l < DEPTH; l++) {
    ln_k<<<TTOK, 256, 0, stream>>>(xf, attn_g + (size_t)l * DIM, hbf);
    gemm_k<0><<<dim3(DIM / 64, TTOK / 128), 256, 0, stream>>>(
        hbf, wq_t + (size_t)l * DIM * DIM, tmpf, TTOK, DIM, DIM, DIM, DIM, DIM);
    gemm_k<0><<<dim3(128 / 64, TTOK / 128), 256, 0, stream>>>(
        hbf, wkv_t + (size_t)l * DIM * 128, kvf, TTOK, 128, DIM, DIM, DIM, 128);
    qpost_k<<<BATCH * NTOKP, 256, 0, stream>>>(tmpf, cosb, sinb, qbf);
    kvpost_k<<<BATCH * NKROWS, 32, 0, stream>>>(kvf, null_kv + (size_t)l * 2 * DH,
                                                cosb, sinb, kbf, vtb);
    attn_k<<<dim3(9, HEADS, BATCH), 256, 0, stream>>>(qbf, kbf, vtb, biasb, aobf);
    gemm_k<0><<<dim3(DIM / 64, TTOK / 128), 256, 0, stream>>>(
        aobf, wo_t + (size_t)l * DIM * DIM, tmpf, TTOK, DIM, DIM, DIM, DIM, DIM);
    ln_res_k<<<TTOK, 256, 0, stream>>>(tmpf, out_g + (size_t)l * DIM, xf);

    ln_k<<<TTOK, 256, 0, stream>>>(xf, ff_g + (size_t)l * DIM, hbf);
    gemm_gated_k<<<dim3(FF / 64, TTOK / 128), 256, 0, stream>>>(
        hbf, w1_t + (size_t)l * DIM * 2 * FF, gbf, TTOK, DIM, DIM, DIM);
    gemm_k<0><<<dim3(DIM / 64, TTOK / 128), 256, 0, stream>>>(
        gbf, w2_t + (size_t)l * FF * DIM, tmpf, TTOK, DIM, FF, FF, FF, DIM);
    add_res_k<<<4096, 256, 0, stream>>>(xf, tmpf, (size_t)TTOK * DIM);
  }

  ln_last_k<<<BATCH, 256, 0, stream>>>(xf, fin_g, hbf);
  gemm_k<0><<<dim3(DIM / 64, 1), 256, 0, stream>>>(
      hbf, proj_t, d_out, BATCH, DIM, DIM, DIM, DIM, DIM);
}
